// GRU_77171972374673
// MI455X (gfx1250) — matrix operand order repaired and checked by reading, not compiled, not run on hardware
//
#include <hip/hip_runtime.h>
#include <math.h>

typedef __attribute__((ext_vector_type(16))) __bf16 v16bf;
typedef __attribute__((ext_vector_type(8)))  __bf16 v8bf;
typedef __attribute__((ext_vector_type(8)))  float  v8f;

constexpr int B_ = 256, T_ = 512, E_ = 10, H_ = 1024, C_ = 10;
constexpr int K_ = H_ + E_;   // 1034
constexpr int KP = 1056;      // padded to 33 * 32
constexpr int KB = KP / 32;   // 33 k-steps of 32

// ---------------------------------------------------------------- zero init
__global__ void k_zero(float* p, int n) {
  int i = blockIdx.x * blockDim.x + threadIdx.x;
  if (i < n) p[i] = 0.f;
}

// ---------------- convert the 3 recurrent weights [H,K] f32 -> [H,KP] bf16
__global__ void k_cvt_w(const float* __restrict__ Wz, const float* __restrict__ Wr,
                        const float* __restrict__ W,
                        __bf16* __restrict__ Wzb, __bf16* __restrict__ Wrb,
                        __bf16* __restrict__ Wb) {
  int i = blockIdx.x * blockDim.x + threadIdx.x;
  const int total = H_ * KP;
  if (i >= 3 * total) return;
  int w = i / total, rem = i - w * total;
  int j = rem / KP, k = rem - j * KP;
  const float* src = (w == 0) ? Wz : (w == 1) ? Wr : W;
  __bf16* dst = (w == 0) ? Wzb : (w == 1) ? Wrb : Wb;
  float v = (k < K_) ? src[j * K_ + k] : 0.f;
  dst[(size_t)j * KP + k] = (__bf16)v;
}

// ---------------------------------------------------------------- helpers
__device__ __forceinline__ v16bf load_afrag(const __bf16* As, int arow, int k0, int ksel) {
  v8bf lo = *(const v8bf*)&As[arow * KP + k0 + ksel * 16];
  v8bf hi = *(const v8bf*)&As[arow * KP + k0 + ksel * 16 + 8];
  v16bf a;
#pragma unroll
  for (int i = 0; i < 8; ++i) { a[i] = lo[i]; a[i + 8] = hi[i]; }
  return a;
}

// --------------------------------- phase 1: z = sig(hx Wz^T), r = sig(hx Wr^T)
__global__ void __launch_bounds__(256)
k_gates(const float* __restrict__ hprev, const int* __restrict__ x,
        const float* __restrict__ embed,
        const __bf16* __restrict__ Wzb, const __bf16* __restrict__ Wrb,
        float* __restrict__ z, float* __restrict__ r, int t) {
  __shared__ __align__(16) __bf16 As[16 * KP];
  const int tid  = threadIdx.x;
  const int mt   = blockIdx.y;            // batch tile (16 rows)
  const int gate = blockIdx.z;            // 0 = z, 1 = r
  const __bf16* __restrict__ Wg = gate ? Wrb : Wzb;
  float* __restrict__ outp = gate ? r : z;

  // stage A = [h | embed[x_t]] as bf16, zero padded to KP
  for (int idx = tid; idx < 16 * KP; idx += 256) {
    int row = idx / KP, k = idx - row * KP;
    int b = mt * 16 + row;
    float v;
    if (k < H_)      v = hprev[b * H_ + k];
    else if (k < K_) v = embed[x[b * T_ + t] * E_ + (k - H_)];
    else             v = 0.f;
    As[idx] = (__bf16)v;
  }
  __syncthreads();

  const int wave = tid >> 5, lane = tid & 31;
  const int l15 = lane & 15, ksel = lane >> 4;
  const int nb0 = blockIdx.x * 256 + wave * 32;   // this wave: 32 output cols

  v8f acc0 = {}, acc1 = {};
  for (int kb = 0; kb < KB; ++kb) {
    const int k0 = kb * 32;
    v16bf a  = load_afrag(As, l15, k0, ksel);
    v16bf b0 = *(const v16bf*)&Wg[(size_t)(nb0 + l15) * KP + k0 + ksel * 16];
    v16bf b1 = *(const v16bf*)&Wg[(size_t)(nb0 + 16 + l15) * KP + k0 + ksel * 16];
    acc0 = __builtin_amdgcn_wmma_f32_16x16x32_bf16(false, a, false, b0, (short)0, acc0, false, false);
    acc1 = __builtin_amdgcn_wmma_f32_16x16x32_bf16(false, a, false, b1, (short)0, acc1, false, false);
  }

#pragma unroll
  for (int v = 0; v < 8; ++v) {
    int m = v + ksel * 8;
    int b = mt * 16 + m;
    outp[b * H_ + nb0 + l15]      = 1.f / (1.f + __expf(-acc0[v]));
    outp[b * H_ + nb0 + 16 + l15] = 1.f / (1.f + __expf(-acc1[v]));
  }
}

// ---------------- phase 2: h_tilde = tanh([h*r | x] W^T); h_new = (1-z)h + z*h_tilde
__global__ void __launch_bounds__(256)
k_update(const float* __restrict__ hprev, float* __restrict__ hnext,
         const int* __restrict__ x, const float* __restrict__ embed,
         const __bf16* __restrict__ Wb,
         const float* __restrict__ z, const float* __restrict__ r, int t) {
  __shared__ __align__(16) __bf16 As[16 * KP];
  const int tid = threadIdx.x;
  const int mt  = blockIdx.y;

  for (int idx = tid; idx < 16 * KP; idx += 256) {
    int row = idx / KP, k = idx - row * KP;
    int b = mt * 16 + row;
    float v;
    if (k < H_)      v = hprev[b * H_ + k] * r[b * H_ + k];
    else if (k < K_) v = embed[x[b * T_ + t] * E_ + (k - H_)];
    else             v = 0.f;
    As[idx] = (__bf16)v;
  }
  __syncthreads();

  const int wave = tid >> 5, lane = tid & 31;
  const int l15 = lane & 15, ksel = lane >> 4;
  const int nb0 = blockIdx.x * 256 + wave * 32;

  v8f acc0 = {}, acc1 = {};
  for (int kb = 0; kb < KB; ++kb) {
    const int k0 = kb * 32;
    v16bf a  = load_afrag(As, l15, k0, ksel);
    v16bf b0 = *(const v16bf*)&Wb[(size_t)(nb0 + l15) * KP + k0 + ksel * 16];
    v16bf b1 = *(const v16bf*)&Wb[(size_t)(nb0 + 16 + l15) * KP + k0 + ksel * 16];
    acc0 = __builtin_amdgcn_wmma_f32_16x16x32_bf16(false, a, false, b0, (short)0, acc0, false, false);
    acc1 = __builtin_amdgcn_wmma_f32_16x16x32_bf16(false, a, false, b1, (short)0, acc1, false, false);
  }

#pragma unroll
  for (int v = 0; v < 8; ++v) {
    int m = v + ksel * 8;
    int b = mt * 16 + m;
    {
      int j = nb0 + l15;
      float zv = z[b * H_ + j], hp = hprev[b * H_ + j];
      hnext[b * H_ + j] = (1.f - zv) * hp + zv * tanhf(acc0[v]);
    }
    {
      int j = nb0 + 16 + l15;
      float zv = z[b * H_ + j], hp = hprev[b * H_ + j];
      hnext[b * H_ + j] = (1.f - zv) * hp + zv * tanhf(acc1[v]);
    }
  }
}

// ---------------------------------------------------------------- projection
__global__ void k_proj(const float* __restrict__ h, const float* __restrict__ Wph,
                       const float* __restrict__ bp, float* __restrict__ out) {
  int b = blockIdx.x;
  int c = threadIdx.x >> 5, lane = threadIdx.x & 31;
  float s = 0.f;
  for (int j = lane; j < H_; j += 32) s += h[b * H_ + j] * Wph[c * H_ + j];
#pragma unroll
  for (int off = 16; off; off >>= 1) s += __shfl_down(s, off, 32);
  if (lane == 0) out[b * C_ + c] = s + bp[c];
}

// ---------------------------------------------------------------- launcher
extern "C" void kernel_launch(void* const* d_in, const int* in_sizes, int n_in,
                              void* d_out, int out_size, void* d_ws, size_t ws_size,
                              hipStream_t stream) {
  const int*   x     = (const int*)d_in[0];
  const float* embed = (const float*)d_in[1];
  const float* Wz    = (const float*)d_in[2];
  const float* Wr    = (const float*)d_in[3];
  const float* W     = (const float*)d_in[4];
  const float* Wph   = (const float*)d_in[5];
  const float* bp    = (const float*)d_in[6];
  float* out = (float*)d_out;

  auto al = [](size_t v) { return (v + 255) & ~size_t(255); };
  char* ws = (char*)d_ws;
  size_t off = 0;
  float*  h0   = (float*)(ws + off);  off = al(off + (size_t)B_ * H_ * 4);
  float*  h1   = (float*)(ws + off);  off = al(off + (size_t)B_ * H_ * 4);
  float*  zbuf = (float*)(ws + off);  off = al(off + (size_t)B_ * H_ * 4);
  float*  rbuf = (float*)(ws + off);  off = al(off + (size_t)B_ * H_ * 4);
  __bf16* Wzb  = (__bf16*)(ws + off); off = al(off + (size_t)H_ * KP * 2);
  __bf16* Wrb  = (__bf16*)(ws + off); off = al(off + (size_t)H_ * KP * 2);
  __bf16* Wb   = (__bf16*)(ws + off); off = al(off + (size_t)H_ * KP * 2);

  k_zero<<<(B_ * H_ + 255) / 256, 256, 0, stream>>>(h0, B_ * H_);
  k_cvt_w<<<(3 * H_ * KP + 255) / 256, 256, 0, stream>>>(Wz, Wr, W, Wzb, Wrb, Wb);

  dim3 gg(H_ / 256, B_ / 16, 2);  // (4, 16, 2)
  dim3 gu(H_ / 256, B_ / 16, 1);  // (4, 16)
  float* bufs[2] = {h0, h1};
  for (int t = 0; t < T_; ++t) {
    float* hp = bufs[t & 1];
    float* hn = bufs[(t + 1) & 1];
    k_gates<<<gg, 256, 0, stream>>>(hp, x, embed, Wzb, Wrb, zbuf, rbuf, t);
    k_update<<<gu, 256, 0, stream>>>(hp, hn, x, embed, Wb, zbuf, rbuf, t);
  }
  // T_ is even -> final hidden state lives in bufs[0] == h0
  k_proj<<<B_, C_ * 32, 0, stream>>>(h0, Wph, bp, out);
}
